// MetaPathConnector_3667902070992
// MI455X (gfx1250) — matrix operand order repaired and checked by reading, not compiled, not run on hardware
//
#include <hip/hip_runtime.h>
#include <hip/hip_bf16.h>
#include <math.h>

typedef __attribute__((ext_vector_type(16))) _Float16 v16h;
typedef __attribute__((ext_vector_type(8)))  _Float16 v8h;
typedef __attribute__((ext_vector_type(8)))  float    v8f;

#define NN   16384
#define DD   128
#define TOPK 11          // K+1 = 11 (top-10 neighbors + self)
#define ROWS 16          // rows per workgroup (one WMMA M-tile)
#define NWAVES 8         // 8 waves -> 128 columns per iteration
#define COLS_ITER (NWAVES * 16)
#define STRENGTH 0.1f
#define NEG_INF  (-3.0e38f)

// ---------------------------------------------------------------------------
// Kernel 1: proj = feat @ W^T  (torch Linear), row L2 norm, f16 normalized copy
// one block (128 threads) per row
// ---------------------------------------------------------------------------
__global__ __launch_bounds__(128) void mpc_proj_kernel(
    const float* __restrict__ feat, const float* __restrict__ W,
    float* __restrict__ proj, _Float16* __restrict__ nrmh)
{
    __shared__ float fr[DD];
    __shared__ float red[DD];
    const int row = blockIdx.x;
    const int t   = threadIdx.x;

    fr[t] = feat[(size_t)row * DD + t];
    __syncthreads();

    // proj[row][t] = sum_k feat[row][k] * W[t][k]   (W row-major [D][D])
    const float* wr = W + (size_t)t * DD;
    float acc = 0.0f;
#pragma unroll 8
    for (int k = 0; k < DD; ++k) acc = fmaf(fr[k], wr[k], acc);

    proj[(size_t)row * DD + t] = acc;

    red[t] = acc * acc;
    __syncthreads();
#pragma unroll
    for (int s = 64; s > 0; s >>= 1) {
        if (t < s) red[t] += red[t + s];
        __syncthreads();
    }
    const float nrm = sqrtf(red[0]);
    const float inv = 1.0f / fmaxf(nrm, 1e-12f);   // torch F.normalize semantics
    nrmh[(size_t)row * DD + t] = (_Float16)(acc * inv);
}

// ---------------------------------------------------------------------------
// Kernel 2: fused  sims = nrm @ nrm^T  (WMMA f16->f32)  +  per-row top-11
//           + self-masked softmax + sparse propagation + residual.
// 256 threads (8 waves) per block; block handles 16 rows.
// Double-buffered sim tile -> single barrier per column iteration.
// ---------------------------------------------------------------------------
__global__ __launch_bounds__(256) void mpc_sim_topk_kernel(
    const float*    __restrict__ proj,
    const _Float16* __restrict__ nrmh,
    const float*    __restrict__ feat,
    const float*    __restrict__ emb,
    float*          __restrict__ out)
{
    __shared__ float S[2][ROWS][COLS_ITER + 4];  // double-buffered sim tile
    __shared__ float mv[ROWS][16 * TOPK];        // partial top-k merge: vals
    __shared__ int   mi[ROWS][16 * TOPK];        // partial top-k merge: idx
    __shared__ float wts[ROWS][TOPK];            // softmax weights
    __shared__ int   wid[ROWS][TOPK];            // neighbor indices

    const int tid  = threadIdx.x;
    const int wave = tid >> 5;
    const int lane = tid & 31;
    const int m0   = blockIdx.x * ROWS;

    // ---- A fragments: 16x128 f16 tile of nrmh rows [m0 .. m0+15]; fixed for block.
    // A 16x32 layout (matches B): lane(0-15)=row M, K elems 0..15;
    // lane(16-31)=row M, K elems 16..31.
    v16h afrag[4];
    {
        const int  l  = lane & 15;
        const int  hi = lane >> 4;
        const _Float16* arow = nrmh + (size_t)(m0 + l) * DD;
#pragma unroll
        for (int s = 0; s < 4; ++s) {
            const int kb = s * 32;
            v8h lo  = *(const v8h*)(arow + kb + hi * 16);
            v8h hi8 = *(const v8h*)(arow + kb + 8 + hi * 16);
#pragma unroll
            for (int e = 0; e < 8; ++e) { afrag[s][e] = lo[e]; afrag[s][8 + e] = hi8[e]; }
        }
    }

    // ---- private top-11 list (sorted descending), 16 threads per row
    float tv[TOPK]; int ti[TOPK];
#pragma unroll
    for (int j = 0; j < TOPK; ++j) { tv[j] = NEG_INF; ti[j] = -1; }

    const int srow = tid >> 4;      // scan row 0..15
    const int ssub = tid & 15;      // scan sub-column group
    const int ccol  = wave * 16 + (lane & 15);  // D-tile scatter column
    const int rbase = (lane >> 4) * 8;          // D-tile scatter row base

    int p = 0;
    for (int cb = 0; cb < NN; cb += COLS_ITER, p ^= 1) {
        // ---- B fragments: columns [n0..n0+15] of sims == rows of nrmh.
        // B 32x16 layout: lanes 0-15 col=n, K kb+0..15; lanes 16-31 col=n, K kb+16..31.
        const int n0  = cb + wave * 16;
        const int col = n0 + (lane & 15);
        const _Float16* brow = nrmh + (size_t)col * DD;
        const int koff = (lane >> 4) * 16;

        v8f c = {0.f, 0.f, 0.f, 0.f, 0.f, 0.f, 0.f, 0.f};
#pragma unroll
        for (int s = 0; s < 4; ++s) {
            v16h b = *(const v16h*)(brow + s * 32 + koff);
            c = __builtin_amdgcn_wmma_f32_16x16x32_f16(
                    false, afrag[s], false, b, (short)0, c, false, false);
        }

        // D layout: VGPR v, lanes 0-15: M=v, N=lane; lanes 16-31: M=v+8, N=lane-16
        // Safe to write S[p] without a pre-barrier: the barrier below in the
        // previous iteration already ordered everyone's scan of S[p] (iter-2)
        // before these writes.
#pragma unroll
        for (int v = 0; v < 8; ++v) S[p][rbase + v][ccol] = c[v];
        __syncthreads();

        // ---- top-k scan: 16 threads per row, 8 columns each (two b128 loads)
        const float4* sp = (const float4*)&S[p][srow][ssub * 8];
        float4 va = sp[0];
        float4 vb = sp[1];
        float vals[8] = {va.x, va.y, va.z, va.w, vb.x, vb.y, vb.z, vb.w};
#pragma unroll
        for (int i = 0; i < 8; ++i) {
            const float v  = vals[i];
            const int   gc = cb + ssub * 8 + i;
            if (v > tv[TOPK - 1]) {
                float cv = v; int ci = gc;
#pragma unroll
                for (int j = 0; j < TOPK; ++j) {
                    if (cv > tv[j]) {
                        float t1 = tv[j]; int t2 = ti[j];
                        tv[j] = cv; ti[j] = ci; cv = t1; ci = t2;
                    }
                }
            }
        }
    }

    // ---- merge 16 partial lists per row
    __syncthreads();
#pragma unroll
    for (int j = 0; j < TOPK; ++j) {
        mv[srow][ssub * TOPK + j] = tv[j];
        mi[srow][ssub * TOPK + j] = ti[j];
    }
    __syncthreads();

    if (tid < ROWS) {
        const int row  = tid;
        const int grow = m0 + row;
        float bv[TOPK]; int bi[TOPK];
#pragma unroll
        for (int j = 0; j < TOPK; ++j) { bv[j] = NEG_INF; bi[j] = -1; }
        for (int j = 0; j < 16 * TOPK; ++j) {
            float cv = mv[row][j]; int ci = mi[row][j];
            if (cv > bv[TOPK - 1]) {
#pragma unroll
                for (int q = 0; q < TOPK; ++q) {
                    if (cv > bv[q]) {
                        float t1 = bv[q]; int t2 = bi[q];
                        bv[q] = cv; bi[q] = ci; cv = t1; ci = t2;
                    }
                }
            }
        }
        // self-masked softmax over the top-11 (reference: mask idx==row)
        float mx = NEG_INF;
#pragma unroll
        for (int j = 0; j < TOPK; ++j)
            if (bi[j] >= 0 && bi[j] != grow) mx = fmaxf(mx, bv[j]);
        float se = 0.0f; float w[TOPK];
#pragma unroll
        for (int j = 0; j < TOPK; ++j) {
            const bool ok = (bi[j] >= 0) && (bi[j] != grow);
            const float e = ok ? expf(bv[j] - mx) : 0.0f;
            w[j] = e; se += e;
        }
        const float inv = 1.0f / se;
#pragma unroll
        for (int j = 0; j < TOPK; ++j) {
            wts[row][j] = w[j] * inv;
            wid[row][j] = (bi[j] < 0) ? 0 : bi[j];
        }
    }
    __syncthreads();

    // ---- sparse propagation + residual: 16 threads/row, 8 dims each
    {
        const int d0 = ssub * 8;
        float acc[8];
#pragma unroll
        for (int i = 0; i < 8; ++i) acc[i] = 0.0f;
#pragma unroll
        for (int j = 0; j < TOPK; ++j) {
            const float wv  = wts[srow][j];
            const int   src = wid[srow][j];
            const float* prow = proj + (size_t)src * DD + d0;
#pragma unroll
            for (int i = 0; i < 8; ++i) acc[i] = fmaf(wv, prow[i], acc[i]);
        }
        const int grow = m0 + srow;
        const float* frow = feat + (size_t)grow * DD + d0;
        float*       orow = out  + (size_t)grow * DD + d0;
#pragma unroll
        for (int i = 0; i < 8; ++i)
            orow[i] = frow[i] + STRENGTH * (acc[i] + emb[d0 + i]);
    }
}

// ---------------------------------------------------------------------------
extern "C" void kernel_launch(void* const* d_in, const int* in_sizes, int n_in,
                              void* d_out, int out_size, void* d_ws, size_t ws_size,
                              hipStream_t stream) {
    const float* feat = (const float*)d_in[0];   // [N, D]
    const float* W    = (const float*)d_in[1];   // [D, D]
    const float* emb  = (const float*)d_in[2];   // [1, D]
    float* out = (float*)d_out;

    // workspace: proj f32 (8 MB) | nrmh f16 (4 MB)
    float*     proj = (float*)d_ws;
    _Float16*  nrmh = (_Float16*)((char*)d_ws + (size_t)NN * DD * sizeof(float));

    mpc_proj_kernel<<<NN, DD, 0, stream>>>(feat, W, proj, nrmh);
    mpc_sim_topk_kernel<<<NN / ROWS, 256, 0, stream>>>(proj, nrmh, feat, emb, out);
}
